// MultiHeadAttention_90142773609159
// MI455X (gfx1250) — matrix operand order repaired and checked by reading, not compiled, not run on hardware
//
#include <hip/hip_runtime.h>
#include <hip/hip_bf16.h>

// ---------------------------------------------------------------------------
// MHA forward for MI455X (gfx1250, wave32, WMMA).
// All matmuls via v_wmma_f32_16x16x32_bf16 (f32 accumulate).
// GEMM stages the A panel in LDS via global_load_async_to_lds_b128 (ASYNCcnt),
// B fragments are register double-buffered to overlap loads with WMMA.
// B=2 T=2048 D=1024 H=16 DK=64 hardcoded (matches reference).
// ---------------------------------------------------------------------------

#define BB  2
#define TT  2048
#define DD  1024
#define HH  16
#define DKK 64
#define MM  (BB * TT)        // 4096 rows
#define N_QKV (3 * DD)       // 3072

typedef __attribute__((ext_vector_type(16))) __bf16 v16bf;
typedef __attribute__((ext_vector_type(8)))  __bf16 v8bf;
typedef __attribute__((ext_vector_type(4)))  __bf16 v4bf;
typedef __attribute__((ext_vector_type(8)))  float  v8f;
typedef __attribute__((ext_vector_type(4)))  float  v4f;

union V16 { v16bf v; v8bf h[2]; };

__device__ __forceinline__ __bf16 f2bf(float f) {
  union { float f; unsigned u; } in; in.f = f;
  unsigned u = in.u;
  u += 0x7FFFu + ((u >> 16) & 1u);           // round-to-nearest-even
  union { unsigned short s; __bf16 b; } out; out.s = (unsigned short)(u >> 16);
  return out.b;
}

__device__ __forceinline__ v8f wmma_bf16(v16bf a, v16bf b, v8f c) {
  // 8 args: (neg_a, A, neg_b, B, c_mod, C, reuse_a, reuse_b)
  return __builtin_amdgcn_wmma_f32_16x16x32_bf16(false, a, false, b,
                                                 (short)0, c, false, false);
}

// Reduce across the 16 lanes holding one C/D row (xor offsets 1,2,4,8 stay
// inside each 16-lane half of the wave32).
__device__ __forceinline__ float rowmax16(float v) {
#pragma unroll
  for (int off = 1; off < 16; off <<= 1) v = fmaxf(v, __shfl_xor(v, off, 32));
  return v;
}
__device__ __forceinline__ float rowsum16(float v) {
#pragma unroll
  for (int off = 1; off < 16; off <<= 1) v += __shfl_xor(v, off, 32);
  return v;
}

// Async global -> LDS copy, 16 bytes per lane (ASYNCcnt tracked).
__device__ __forceinline__ void async_copy_b128(unsigned lds_off, const void* g) {
  asm volatile("global_load_async_to_lds_b128 %0, %1, off"
               :: "v"(lds_off), "v"(g) : "memory");
}
__device__ __forceinline__ void wait_async0() {
  asm volatile("s_wait_asynccnt 0x0" ::: "memory");
}

// B-fragment quad load: 4 columns tiles x contiguous 16-elt K chunk per lane.
__device__ __forceinline__ void load_b4(const __bf16* __restrict__ Bm, int nbase,
                                        int ln, int c16, int K, int k,
                                        V16 (&ub)[4]) {
#pragma unroll
  for (int n = 0; n < 4; ++n) {
    const __bf16* bptr = Bm + (size_t)(nbase + n * 16 + ln) * K + k + c16;
    ub[n].h[0] = *(const v8bf*)(bptr);
    ub[n].h[1] = *(const v8bf*)(bptr + 8);
  }
}

__device__ __forceinline__ v16bf a_frag(const __bf16* arow, int k, int c8) {
  V16 u;
  u.h[0] = *(const v8bf*)(arow + k + 2 * c8);
  u.h[1] = *(const v8bf*)(arow + k + 2 * c8 + 8);
  return u.v;
}

// ---------------------------------------------------------------------------
// f32 -> bf16 conversion, 4 elements/thread (b128 load, b64 store)
// ---------------------------------------------------------------------------
__global__ __launch_bounds__(256)
void mha_cvt_bf16(const float* __restrict__ in, __bf16* __restrict__ out, int n4) {
  int i = blockIdx.x * 256 + threadIdx.x;
  if (i < n4) {
    v4f f = ((const v4f*)in)[i];
    v4bf b;
#pragma unroll
    for (int j = 0; j < 4; ++j) b[j] = f2bf(f[j]);
    ((v4bf*)out)[i] = b;
  }
}

// ---------------------------------------------------------------------------
// C[M,N] = A[M,K] @ B[N,K]^T + bias, bf16 in, f32 out.
// Block (256 thr, 8 waves) -> 16 rows x 512 cols.  A panel (16 x K) staged in
// LDS once per block via async loads; B fragments double-buffered in regs.
// K must be a multiple of 64.  grid = (M/16, N/512).
// ---------------------------------------------------------------------------
__global__ __launch_bounds__(256)
void mha_gemm_bf16(const __bf16* __restrict__ A, const __bf16* __restrict__ Bm,
                   const float* __restrict__ bias, float* __restrict__ C,
                   int M, int N, int K) {
  __shared__ __bf16 apanel[16 * 1024];          // 32 KB (K <= 1024)

  const int tid  = threadIdx.x;
  const int lane = tid & 31;
  const int wave = tid >> 5;
  const int hl = lane >> 4;
  const int ln = lane & 15;
  const int c8  = hl << 3;
  const int c16 = hl << 4;

  const int mbase = blockIdx.x * 16;
  const int nbase = (blockIdx.y * 8 + wave) * 64;

  // ---- cooperative async copy of A panel (rows mbase..mbase+15, all K) ----
  {
    const char* gsrc = (const char*)(A + (size_t)mbase * K);
    const unsigned nbytes = (unsigned)(16 * K * 2);
    const unsigned ldsbase = (unsigned)(size_t)(&apanel[0]);
    for (unsigned off = (unsigned)tid * 16u; off < nbytes; off += 256u * 16u)
      async_copy_b128(ldsbase + off, gsrc + off);
    wait_async0();
    __syncthreads();
  }

  v8f acc[4] = {v8f{}, v8f{}, v8f{}, v8f{}};
  const __bf16* arow = apanel + (size_t)ln * K;  // LDS row for this lane

  V16 b0[4], b1[4];
  load_b4(Bm, nbase, ln, c16, K, 0, b0);
  for (int k = 0; k < K; k += 64) {
    load_b4(Bm, nbase, ln, c16, K, k + 32, b1);
    v16bf ua0 = a_frag(arow, k, c8);
#pragma unroll
    for (int n = 0; n < 4; ++n) acc[n] = wmma_bf16(ua0, b0[n].v, acc[n]);
    if (k + 64 < K) load_b4(Bm, nbase, ln, c16, K, k + 64, b0);
    v16bf ua1 = a_frag(arow, k + 32, c8);
#pragma unroll
    for (int n = 0; n < 4; ++n) acc[n] = wmma_bf16(ua1, b1[n].v, acc[n]);
  }

  const int rbase = mbase + (hl << 3);
#pragma unroll
  for (int n = 0; n < 4; ++n) {
    const int col = nbase + n * 16 + ln;
    const float bv = bias ? bias[col] : 0.0f;
#pragma unroll
    for (int r = 0; r < 8; ++r)
      C[(size_t)(rbase + r) * N + col] = acc[n][r] + bv;
  }
}

// ---------------------------------------------------------------------------
// RoPE + split qkv(f32, [M, 3D]) into bf16 Q,K (b,h,t,dk) and V^T (b,h,dk,t).
// One thread per (b,t,h,dk-pair).
// ---------------------------------------------------------------------------
__global__ __launch_bounds__(256)
void mha_rope_split(const float* __restrict__ qkv,
                    const float* __restrict__ fcos, const float* __restrict__ fsin,
                    __bf16* __restrict__ Q, __bf16* __restrict__ Kt,
                    __bf16* __restrict__ Vt) {
  int g = blockIdx.x * 256 + threadIdx.x;
  if (g >= BB * TT * HH * (DKK / 2)) return;
  const int d2 = g & 31;  g >>= 5;
  const int h  = g & 15;  g >>= 4;
  const int t  = g & 2047;
  const int b  = g >> 11;

  const size_t row = (size_t)(b * TT + t) * N_QKV;
  const int colq = h * DKK + 2 * d2;
  const float qre = qkv[row + colq],          qim = qkv[row + colq + 1];
  const float kre = qkv[row + DD + colq],     kim = qkv[row + DD + colq + 1];
  const float v0  = qkv[row + 2 * DD + colq], v1  = qkv[row + 2 * DD + colq + 1];
  const float c = fcos[t * 32 + d2], s = fsin[t * 32 + d2];

  const size_t bh = (size_t)(b * HH + h);
  const size_t qk_idx = (bh * TT + t) * DKK + 2 * d2;
  Q[qk_idx]      = f2bf(qre * c - qim * s);
  Q[qk_idx + 1]  = f2bf(qre * s + qim * c);
  Kt[qk_idx]     = f2bf(kre * c - kim * s);
  Kt[qk_idx + 1] = f2bf(kre * s + kim * c);
  Vt[(bh * DKK + 2 * d2) * TT + t]     = f2bf(v0);   // V^T: (b,h,dk,t)
  Vt[(bh * DKK + 2 * d2 + 1) * TT + t] = f2bf(v1);
}

// ---------------------------------------------------------------------------
// Causal flash attention. Q,K: (B*H, T, 64) bf16; Vt: (B*H, 64, T) bf16.
// One wave -> 16-row query tile; 32 keys/iter: 4 WMMA QK^T, online softmax,
// P via LDS (s_wait_dscnt), 4 WMMA PV.  grid = (T/128, B*H), block = 256.
// ---------------------------------------------------------------------------
__global__ __launch_bounds__(256)
void mha_flash_attn(const __bf16* __restrict__ Q, const __bf16* __restrict__ Kt,
                    const __bf16* __restrict__ Vt, __bf16* __restrict__ O) {
  const int lane = threadIdx.x & 31;
  const int wave = threadIdx.x >> 5;
  const int hl = lane >> 4;
  const int ln = lane & 15;
  const int c8 = hl << 3, c16 = hl << 4;

  const int bh = blockIdx.y;                    // b*H + h
  const int qbase = blockIdx.x * 128 + wave * 16;

  const __bf16* Qb = Q  + (size_t)bh * TT * DKK;
  const __bf16* Kb = Kt + (size_t)bh * TT * DKK;
  const __bf16* Vb = Vt + (size_t)bh * DKK * TT;

  __shared__ __bf16 pshare[8][16 * 32];         // per-wave P staging (8 KB)
  __bf16* pt = &pshare[wave][0];

  // Q A-fragments for dk 0..31 / 32..63 (row = qbase + ln)
  const __bf16* qrow = Qb + (size_t)(qbase + ln) * DKK;
  V16 qa0, qa1;
  qa0.h[0] = *(const v8bf*)(qrow + c16);
  qa0.h[1] = *(const v8bf*)(qrow + c16 + 8);
  qa1.h[0] = *(const v8bf*)(qrow + 32 + c16);
  qa1.h[1] = *(const v8bf*)(qrow + 40 + c16);

  v8f o[4] = {v8f{}, v8f{}, v8f{}, v8f{}};      // 16 x 64 output accumulator
  float m[8], l[8];
#pragma unroll
  for (int r = 0; r < 8; ++r) { m[r] = -3.0e38f; l[r] = 0.0f; }
  const float scale = 0.125f;                   // 1/sqrt(64)

  for (int kb = 0; kb <= qbase + 15; kb += 32) {   // wave-uniform causal range
    // prefetch next iteration's K/V tiles (global_prefetch_b8)
    if (kb + 32 <= qbase + 15) {
      __builtin_prefetch(Kb + (size_t)(kb + 32 + lane) * DKK, 0, 3);
      __builtin_prefetch(Vb + (size_t)lane * TT + kb + 32, 0, 3);
      __builtin_prefetch(Vb + (size_t)(lane + 32) * TT + kb + 32, 0, 3);
    }

    // ---- S = Q K^T (16 x 32) ----
    v8f s0 = v8f{}, s1 = v8f{};
#pragma unroll
    for (int f = 0; f < 2; ++f) {
      const __bf16* kp0 = Kb + (size_t)(kb + ln) * DKK + f * 32 + c16;
      const __bf16* kp1 = Kb + (size_t)(kb + 16 + ln) * DKK + f * 32 + c16;
      V16 b0, b1;
      b0.h[0] = *(const v8bf*)(kp0); b0.h[1] = *(const v8bf*)(kp0 + 8);
      b1.h[0] = *(const v8bf*)(kp1); b1.h[1] = *(const v8bf*)(kp1 + 8);
      s0 = wmma_bf16(f ? qa1.v : qa0.v, b0.v, s0);
      s1 = wmma_bf16(f ? qa1.v : qa0.v, b1.v, s1);
    }

    // ---- online softmax (rows in VGPR index, cols across 16 lanes) ----
#pragma unroll
    for (int r = 0; r < 8; ++r) {
      const int row = qbase + r + (hl << 3);
      const int col0 = kb + ln, col1 = kb + 16 + ln;
      const float v0 = (col0 <= row) ? s0[r] * scale : -3.0e38f;
      const float v1 = (col1 <= row) ? s1[r] * scale : -3.0e38f;
      const float mnew = fmaxf(m[r], rowmax16(fmaxf(v0, v1)));
      const float corr = __expf(m[r] - mnew);
      const float p0 = (col0 <= row) ? __expf(v0 - mnew) : 0.0f;
      const float p1 = (col1 <= row) ? __expf(v1 - mnew) : 0.0f;
      l[r] = l[r] * corr + rowsum16(p0 + p1);
      m[r] = mnew;
      o[0][r] *= corr; o[1][r] *= corr; o[2][r] *= corr; o[3][r] *= corr;
      pt[(r + (hl << 3)) * 32 + ln]      = f2bf(p0);
      pt[(r + (hl << 3)) * 32 + 16 + ln] = f2bf(p1);
    }
    // wave-synchronous LDS exchange: C-layout -> A-layout
    asm volatile("s_wait_dscnt 0x0" ::: "memory");

    const __bf16* prow = pt + (size_t)ln * 32;
    V16 pa;
    pa.h[0] = *(const v8bf*)(prow + c16);
    pa.h[1] = *(const v8bf*)(prow + c16 + 8);

    // ---- O += P V  (V^T layout -> contiguous B-fragments) ----
#pragma unroll
    for (int n = 0; n < 4; ++n) {
      const __bf16* vp = Vb + (size_t)(n * 16 + ln) * TT + kb + c16;
      V16 vb;
      vb.h[0] = *(const v8bf*)(vp); vb.h[1] = *(const v8bf*)(vp + 8);
      o[n] = wmma_bf16(pa.v, vb.v, o[n]);
    }
  }

  // epilogue: normalize, store bf16 attention output (B*T, D)
  const int b = bh >> 4, h = bh & 15;
#pragma unroll
  for (int r = 0; r < 8; ++r) {
    const int t = qbase + r + (hl << 3);
    const float inv = 1.0f / l[r];
#pragma unroll
    for (int n = 0; n < 4; ++n)
      O[(size_t)(b * TT + t) * DD + h * DKK + n * 16 + ln] = f2bf(o[n][r] * inv);
  }
}

// ---------------------------------------------------------------------------
// Host launch
// ---------------------------------------------------------------------------
extern "C" void kernel_launch(void* const* d_in, const int* in_sizes, int n_in,
                              void* d_out, int out_size, void* d_ws, size_t ws_size,
                              hipStream_t stream) {
  (void)in_sizes; (void)n_in; (void)out_size; (void)ws_size;
  const float* x      = (const float*)d_in[0];
  const float* w_qkv  = (const float*)d_in[1];
  const float* b_qkv  = (const float*)d_in[2];
  const float* w_out  = (const float*)d_in[3];
  const float* b_out  = (const float*)d_in[4];
  const float* fcos   = (const float*)d_in[5];
  const float* fsin   = (const float*)d_in[6];
  float* out = (float*)d_out;

  char* ws = (char*)d_ws;
  float*  qkvf  = (float*) (ws + 0);          // 4096*3072 f32   = 50331648 B
  __bf16* xb    = (__bf16*)(ws + 50331648);   // 4096*1024 bf16  =  8388608 B
  __bf16* wqkvb = (__bf16*)(ws + 58720256);   // 3072*1024 bf16  =  6291456 B
  __bf16* woutb = (__bf16*)(ws + 65011712);   // 1024*1024 bf16  =  2097152 B
  __bf16* qb    = (__bf16*)(ws + 67108864);   // (B,H,T,DK) bf16 =  8388608 B
  __bf16* kb    = (__bf16*)(ws + 75497472);   // (B,H,T,DK) bf16 =  8388608 B
  __bf16* vtb   = (__bf16*)(ws + 83886080);   // (B,H,DK,T) bf16 =  8388608 B
  __bf16* attnb = (__bf16*)(ws + 92274688);   // (B*T, D)   bf16 =  8388608 B

  // 1) downconvert inputs to bf16 (4 elems/thread)
  mha_cvt_bf16<<<(MM * DD / 4 + 255) / 256, 256, 0, stream>>>(x, xb, MM * DD / 4);
  mha_cvt_bf16<<<(N_QKV * DD / 4 + 255) / 256, 256, 0, stream>>>(w_qkv, wqkvb, N_QKV * DD / 4);
  mha_cvt_bf16<<<(DD * DD / 4 + 255) / 256, 256, 0, stream>>>(w_out, woutb, DD * DD / 4);

  // 2) qkv = x @ w_qkv^T + b_qkv   (4096 x 3072 x 1024)
  mha_gemm_bf16<<<dim3(MM / 16, N_QKV / 512), 256, 0, stream>>>(
      xb, wqkvb, b_qkv, qkvf, MM, N_QKV, DD);

  // 3) RoPE + split into Q, K, V^T (bf16)
  mha_rope_split<<<(BB * TT * HH * 32 + 255) / 256, 256, 0, stream>>>(
      qkvf, fcos, fsin, qb, kb, vtb);

  // 4) causal flash attention
  mha_flash_attn<<<dim3(TT / 128, BB * HH), 256, 0, stream>>>(qb, kb, vtb, attnb);

  // 5) out = attn @ w_out^T + b_out   (4096 x 1024 x 1024) -> f32 d_out
  mha_gemm_bf16<<<dim3(MM / 16, DD / 512), 256, 0, stream>>>(
      attnb, woutb, b_out, out, MM, DD, DD);
}
